// Contextual_MFN_83081847374144
// MI455X (gfx1250) — matrix operand order repaired and checked by reading, not compiled, not run on hardware
//
#include <hip/hip_runtime.h>
#include <stdint.h>
#include <math.h>

// ---------------------------------------------------------------------------
// Contextual MFN scan for gfx1250 (MI455X).
// Persistent-workgroup design: 16 blocks x 256 threads (8 wave32's).
// Block b owns batch rows [16b, 16b+16) and runs the full T=512 scan.
// GEMMs via v_wmma_f32_16x16x32_bf16; weights pre-packed to bf16 in d_ws.
// All global accesses use explicit addrspace(1) casts so codegen emits
// global_load_* (LOADcnt only) instead of flat_load_* (LOADcnt+DScnt).
// ---------------------------------------------------------------------------

typedef __bf16 bf16_t;
typedef __attribute__((ext_vector_type(16))) __bf16 v16bf;
typedef __attribute__((ext_vector_type(8)))  float  v8f;
typedef __attribute__((ext_vector_type(4)))  unsigned int u32x4;  // trivially copyable 128b

typedef __attribute__((address_space(1))) u32x4  gu32x4;
typedef __attribute__((address_space(1))) float  gfloat;
typedef __attribute__((address_space(1))) bf16_t gbf16;

__device__ __forceinline__ u32x4 ldg_u4(const void* p) {
  return *(const gu32x4*)p;
}
__device__ __forceinline__ float ldg_f(const float* p) {
  return *(const gfloat*)p;
}
__device__ __forceinline__ void stg_f(float* p, float v) {
  *(gfloat*)p = v;
}

#define T_STEPS 512
#define NBATCH  256
#define DL 300
#define DA 74
#define DV 35
#define DX 409          // D_L + D_A + D_V
#define DHID 128
#define MEMD 256

// --- d_ws byte offsets for packed bf16 weights -----------------------------
#define OFF_WL   0u          // 512 x 448  bf16  (Wih_l | Whh_l | pad)
#define OFF_WA   458752u     // 512 x 224  bf16
#define OFF_WV   688128u     // 512 x 192  bf16
#define OFF_BL   884736u     // 512 f32 (bih+bhh)
#define OFF_BA   886784u
#define OFF_BV   888832u
#define OFF_A11  890880u     // 256 x 768
#define OFF_A12  1284096u    // 768 x 256
#define OFF_A21  1677312u    // 256 x 768
#define OFF_A22  2070528u    // 256 x 256
#define OFF_G11  2201600u    // 256 x 1024
#define OFF_G12  2725888u    // 256 x 256
#define OFF_G21  2856960u    // 256 x 1024
#define OFF_G22  3381248u    // 256 x 256
#define OFF_OW1  3512320u    // 256 x 640
#define WS_END   3840000u

// --- LDS layout (bytes) ----------------------------------------------------
#define L_C     0u           // float [3][16][128]  cell states
#define L_H     24576u       // bf16  [3][16][128]  hidden states
#define L_MEM   36864u       // float [16][256]     memory
#define L_CS    53248u       // bf16  [16][1024]    cStar / attended / both
#define L_AB    86016u       // bf16  [16][640]     A-staging (gates, Y1, last_hs)
#define L_GF    106496u      // float [16][768]     att logits / out Y
#define L_SCH   155648u      // bf16  [16][256]     cHat
#define L_SG1   163840u      // bf16  [16][256]     gamma1
#define L_SG2   172032u      // bf16  [16][256]     gamma2
#define SMEM_BYTES 180224u

// ---------------------------------------------------------------------------
__device__ __forceinline__ bf16_t to_bf16(float f) {
  unsigned u = __builtin_bit_cast(unsigned, f);
  u += 0x7fffu + ((u >> 16) & 1u);           // round-to-nearest-even
  unsigned short h = (unsigned short)(u >> 16);
  return __builtin_bit_cast(bf16_t, h);
}
__device__ __forceinline__ float from_bf16(bf16_t b) {
  unsigned short h = __builtin_bit_cast(unsigned short, b);
  unsigned u = ((unsigned)h) << 16;
  return __builtin_bit_cast(float, u);
}
__device__ __forceinline__ float sigf(float x) { return 1.0f / (1.0f + __expf(-x)); }
__device__ __forceinline__ v8f vzero8() {
  v8f z = {0.f,0.f,0.f,0.f,0.f,0.f,0.f,0.f};
  return z;
}

// A fragment: 16x32 bf16 tile from LDS (row-major, stride sA elements).
// ISA 7.12.2 layout: lanes 0-15 hold K=k0+[0..7],[16..23]; lanes 16-31 hold
// K=k0+[8..15],[24..31]; M = lane & 15.
__device__ __forceinline__ v16bf load_a_frag(const bf16_t* A, int sA, int lane, int k0) {
  int m = lane & 15, sel = lane >> 4;
  const bf16_t* p = A + m * sA + k0 + sel * 8;
  v16bf a;
  u32x4* d = reinterpret_cast<u32x4*>(&a);
  d[0] = *reinterpret_cast<const u32x4*>(p);
  d[1] = *reinterpret_cast<const u32x4*>(p + 16);
  return a;
}

// B fragment: 32x16 bf16 tile from GLOBAL. B[k][n] = W[n][k] with W row-major
// (Nout x K). Lane L: column n0 + (L&15); lanes 0-15 K=[0..7],[16..23], lanes 16-31 K=[8..15],[24..31].
__device__ __forceinline__ v16bf load_b_frag(const bf16_t* W, int sB, int lane, int k0, int n0) {
  int n = n0 + (lane & 15), sel = lane >> 4;
  const bf16_t* p = W + (size_t)n * sB + k0 + sel * 8;
  v16bf b;
  u32x4* d = reinterpret_cast<u32x4*>(&b);
  d[0] = ldg_u4(p);
  d[1] = ldg_u4(p + 16);
  return b;
}

enum { ACT_RELU_BF = 0, ACT_F32 = 1, ACT_TANH_BF = 2, ACT_SIG_BF = 3 };

// One dense layer: out[16 x Nout] = act(A[16 x K] @ W^T + bias).
// NT = output tiles per wave (Nout = NT*128). K-outer / multi-accumulator so
// one LDS A-fragment feeds NT WMMAs.
template <int NT, int ACT>
__device__ __forceinline__ void layer(const bf16_t* A, int sA, int K,
                                      const bf16_t* W, const float* bias,
                                      bf16_t* dstb, float* dstf, int sD,
                                      int wave, int lane) {
  v8f acc[NT];
#pragma unroll
  for (int i = 0; i < NT; ++i) acc[i] = vzero8();
  for (int k0 = 0; k0 < K; k0 += 32) {
    v16bf a = load_a_frag(A, sA, lane, k0);
#pragma unroll
    for (int i = 0; i < NT; ++i) {
      v16bf b = load_b_frag(W, K, lane, k0, (wave + 8 * i) * 16);
      acc[i] = __builtin_amdgcn_wmma_f32_16x16x32_bf16(false, a, false, b, (short)0,
                                                       acc[i], false, false);
    }
  }
  int n16 = lane & 15, sel = lane >> 4;
#pragma unroll
  for (int i = 0; i < NT; ++i) {
    int col = (wave + 8 * i) * 16 + n16;
    float bc = ldg_f(bias + col);
#pragma unroll
    for (int r = 0; r < 8; ++r) {
      int m = r + sel * 8;
      float v = acc[i][r] + bc;
      if (ACT == ACT_RELU_BF)      dstb[m * sD + col] = to_bf16(fmaxf(v, 0.f));
      else if (ACT == ACT_F32)     dstf[m * sD + col] = v;
      else if (ACT == ACT_TANH_BF) dstb[m * sD + col] = to_bf16(tanhf(v));
      else                         dstb[m * sD + col] = to_bf16(sigf(v));
    }
  }
}

// ---------------------------------------------------------------------------
__global__ void __launch_bounds__(256) mfn_scan_kernel(
    const float* __restrict__ x,
    const bf16_t* WL, const bf16_t* WA, const bf16_t* WV,
    const float* BL, const float* BA, const float* BV,
    const bf16_t* A11, const float* b11, const bf16_t* A12, const float* b12,
    const bf16_t* A21, const float* b21, const bf16_t* A22, const float* b22,
    const bf16_t* G11, const float* bg11, const bf16_t* G12, const float* bg12,
    const bf16_t* G21, const float* bg21, const bf16_t* G22, const float* bg22,
    const bf16_t* OW1, const float* bo1, const float* OW2, const float* bo2,
    float* __restrict__ out) {
  extern __shared__ char smem[];
  float*  st_c   = (float*)(smem + L_C);
  bf16_t* st_h   = (bf16_t*)(smem + L_H);
  float*  st_mem = (float*)(smem + L_MEM);
  bf16_t* CS     = (bf16_t*)(smem + L_CS);
  bf16_t* AB     = (bf16_t*)(smem + L_AB);
  float*  Gf     = (float*)(smem + L_GF);
  bf16_t* Sch    = (bf16_t*)(smem + L_SCH);
  bf16_t* Sg1    = (bf16_t*)(smem + L_SG1);
  bf16_t* Sg2    = (bf16_t*)(smem + L_SG2);

  const int tid = threadIdx.x;
  const int wave = tid >> 5, lane = tid & 31;
  const int r0 = blockIdx.x * 16;

  // zero-init recurrent state (reference ignores the *_prior inputs)
  for (int i = tid; i < 3 * 16 * DHID; i += 256) { st_c[i] = 0.f; st_h[i] = to_bf16(0.f); }
  for (int i = tid; i < 16 * MEMD; i += 256) st_mem[i] = 0.f;
  __syncthreads();

  const int din[3] = {DL, DA, DV};
  const int kp[3]  = {448, 224, 192};
  const int xoff[3] = {0, DL, DL + DA};
  const bf16_t* Wg[3] = {WL, WA, WV};
  const float*  Bg[3] = {BL, BA, BV};

  for (int t = 0; t < T_STEPS; ++t) {
    // prefetch next timestep's x rows into cache (global_prefetch_b8)
    if (t + 1 < T_STEPS) {
      const float* nx = x + ((size_t)(t + 1) * NBATCH + r0) * DX;
      for (int i = tid * 32; i < 16 * DX; i += 256 * 32) __builtin_prefetch(nx + i, 0, 1);
    }

    // 1. cStar low half = prev cell states (bf16)
    for (int i = tid; i < 16 * 384; i += 256) {
      int m = i / 384, k = i % 384;
      CS[m * 1024 + k] = to_bf16(st_c[(k >> 7) * 2048 + m * DHID + (k & 127)]);
    }
    __syncthreads();

    // 2. three LSTM cells (K fused: [x_mod | h_mod], gates via WMMA)
    for (int mod = 0; mod < 3; ++mod) {
      const int dn = din[mod], Kp = kp[mod], xo = xoff[mod];
      // stage A = [x_t | h] bf16, zero-padded to Kp
      for (int i = tid; i < 16 * Kp; i += 256) {
        int m = i / Kp, k = i % Kp;
        bf16_t v;
        if (k < dn)             v = to_bf16(ldg_f(&x[((size_t)t * NBATCH + r0 + m) * DX + xo + k]));
        else if (k < dn + DHID) v = st_h[mod * 2048 + m * DHID + (k - dn)];
        else                    v = to_bf16(0.f);
        AB[m * Kp + k] = v;
      }
      __syncthreads();

      // wave w owns i/f/g/o gate tiles for columns [16w, 16w+16)
      v8f acc[4];
#pragma unroll
      for (int g = 0; g < 4; ++g) acc[g] = vzero8();
      for (int k0 = 0; k0 < Kp; k0 += 32) {
        v16bf a = load_a_frag(AB, Kp, lane, k0);
#pragma unroll
        for (int g = 0; g < 4; ++g) {
          v16bf b = load_b_frag(Wg[mod], Kp, lane, k0, g * DHID + wave * 16);
          acc[g] = __builtin_amdgcn_wmma_f32_16x16x32_bf16(false, a, false, b, (short)0,
                                                           acc[g], false, false);
        }
      }
      // in-register LSTM elementwise on C fragments
      {
        int n16 = lane & 15, sel = lane >> 4;
        int col = wave * 16 + n16;
        const float* bb = Bg[mod];
        float bi = ldg_f(bb + col), bf = ldg_f(bb + DHID + col);
        float bg = ldg_f(bb + 2 * DHID + col), bo = ldg_f(bb + 3 * DHID + col);
#pragma unroll
        for (int r = 0; r < 8; ++r) {
          int m = r + sel * 8;
          float iv = sigf(acc[0][r] + bi);
          float fv = sigf(acc[1][r] + bf);
          float gv = tanhf(acc[2][r] + bg);
          float ov = sigf(acc[3][r] + bo);
          float cn = fv * st_c[mod * 2048 + m * DHID + col] + iv * gv;
          st_c[mod * 2048 + m * DHID + col] = cn;
          CS[m * 1024 + 384 + mod * DHID + col] = to_bf16(cn);
          st_h[mod * 2048 + m * DHID + col] = to_bf16(ov * tanhf(cn));
        }
      }
      __syncthreads();
    }

    // 3. attention MLP
    layer<2, ACT_RELU_BF>(CS, 1024, 768, A11, b11, AB, nullptr, 256, wave, lane);
    __syncthreads();
    layer<6, ACT_F32>(AB, 256, 256, A12, b12, nullptr, Gf, 768, wave, lane);
    __syncthreads();

    // 4. softmax over 768 features, attended = att * cStar, append old mem
    {
      int row = tid >> 4, sub = tid & 15;
      float mx = -3.4e38f;
      for (int j = 0; j < 48; ++j) mx = fmaxf(mx, Gf[row * 768 + sub + 16 * j]);
      for (int off = 8; off; off >>= 1) mx = fmaxf(mx, __shfl_xor(mx, off, 16));
      float sm = 0.f;
      for (int j = 0; j < 48; ++j) sm += __expf(Gf[row * 768 + sub + 16 * j] - mx);
      for (int off = 8; off; off >>= 1) sm += __shfl_xor(sm, off, 16);
      float inv = 1.0f / sm;
      for (int j = 0; j < 48; ++j) {
        int c = sub + 16 * j;
        float a = __expf(Gf[row * 768 + c] - mx) * inv;
        CS[row * 1024 + c] = to_bf16(a * from_bf16(CS[row * 1024 + c]));
      }
      for (int j = 0; j < 16; ++j) {
        int c = sub + 16 * j;
        CS[row * 1024 + 768 + c] = to_bf16(st_mem[row * MEMD + c]);
      }
    }
    __syncthreads();

    // 5. cHat = tanh(MLP(attended))
    layer<2, ACT_RELU_BF>(CS, 1024, 768, A21, b21, AB, nullptr, 256, wave, lane);
    __syncthreads();
    layer<2, ACT_TANH_BF>(AB, 256, 256, A22, b22, Sch, nullptr, 256, wave, lane);
    __syncthreads();

    // 6. gamma1 / gamma2 = sigmoid(MLP(both)), both = [attended | mem] (K=1024)
    layer<2, ACT_RELU_BF>(CS, 1024, 1024, G11, bg11, AB, nullptr, 256, wave, lane);
    __syncthreads();
    layer<2, ACT_SIG_BF>(AB, 256, 256, G12, bg12, Sg1, nullptr, 256, wave, lane);
    __syncthreads();
    layer<2, ACT_RELU_BF>(CS, 1024, 1024, G21, bg21, AB, nullptr, 256, wave, lane);
    __syncthreads();
    layer<2, ACT_SIG_BF>(AB, 256, 256, G22, bg22, Sg2, nullptr, 256, wave, lane);
    __syncthreads();

    // 7. mem = gamma1 * mem + gamma2 * cHat
    for (int i = tid; i < 16 * MEMD; i += 256) {
      st_mem[i] = from_bf16(Sg1[i]) * st_mem[i] + from_bf16(Sg2[i]) * from_bf16(Sch[i]);
    }
    __syncthreads();
  }

  // 8. output head: last_hs = [h_l, h_a, h_v, mem] (16 x 640)
  for (int i = tid; i < 16 * 640; i += 256) {
    int m = i / 640, k = i % 640;
    bf16_t v;
    if (k < 384) v = st_h[(k >> 7) * 2048 + m * DHID + (k & 127)];
    else         v = to_bf16(st_mem[m * MEMD + (k - 384)]);
    AB[m * 640 + k] = v;
  }
  __syncthreads();
  layer<2, ACT_F32>(AB, 640, 640, OW1, bo1, nullptr, Gf, 256, wave, lane);
  __syncthreads();
  {
    int row = tid >> 4, sub = tid & 15;
    float s = 0.f;
    for (int j = 0; j < 16; ++j) {
      int c = sub + 16 * j;
      s += fmaxf(Gf[row * 256 + c], 0.f) * ldg_f(OW2 + c);
    }
    for (int off = 8; off; off >>= 1) s += __shfl_xor(s, off, 16);
    if (sub == 0) stg_f(&out[r0 + row], s + ldg_f(bo2));
  }
}

// ---------------------------------------------------------------------------
// Prep kernels: f32 -> packed bf16 weights in d_ws.
__global__ void pack_gate(const float* __restrict__ Wih, const float* __restrict__ Whh,
                          const float* __restrict__ bih, const float* __restrict__ bhh,
                          bf16_t* __restrict__ Wd, float* __restrict__ bd,
                          int din, int Kp) {
  int total = 512 * Kp;
  for (int i = blockIdx.x * blockDim.x + threadIdx.x; i < total;
       i += gridDim.x * blockDim.x) {
    int o = i / Kp, k = i % Kp;
    float v = 0.f;
    if (k < din)             v = ldg_f(Wih + o * din + k);
    else if (k < din + DHID) v = ldg_f(Whh + o * DHID + (k - din));
    *(gbf16*)(Wd + i) = to_bf16(v);
  }
  for (int o = blockIdx.x * blockDim.x + threadIdx.x; o < 512;
       o += gridDim.x * blockDim.x)
    stg_f(bd + o, ldg_f(bih + o) + ldg_f(bhh + o));
}

__global__ void pack_w(const float* __restrict__ W, bf16_t* __restrict__ Wd, int total) {
  for (int i = blockIdx.x * blockDim.x + threadIdx.x; i < total;
       i += gridDim.x * blockDim.x)
    *(gbf16*)(Wd + i) = to_bf16(ldg_f(W + i));
}

// ---------------------------------------------------------------------------
extern "C" void kernel_launch(void* const* d_in, const int* in_sizes, int n_in,
                              void* d_out, int out_size, void* d_ws, size_t ws_size,
                              hipStream_t stream) {
  (void)in_sizes; (void)n_in; (void)out_size; (void)ws_size;
  const float* x      = (const float*)d_in[0];
  const float* Wih_l  = (const float*)d_in[5];
  const float* Whh_l  = (const float*)d_in[6];
  const float* bih_l  = (const float*)d_in[7];
  const float* bhh_l  = (const float*)d_in[8];
  const float* Wih_a  = (const float*)d_in[9];
  const float* Whh_a  = (const float*)d_in[10];
  const float* bih_a  = (const float*)d_in[11];
  const float* bhh_a  = (const float*)d_in[12];
  const float* Wih_v  = (const float*)d_in[13];
  const float* Whh_v  = (const float*)d_in[14];
  const float* bih_v  = (const float*)d_in[15];
  const float* bhh_v  = (const float*)d_in[16];
  const float* att1W1 = (const float*)d_in[17];
  const float* att1b1 = (const float*)d_in[18];
  const float* att1W2 = (const float*)d_in[19];
  const float* att1b2 = (const float*)d_in[20];
  const float* att2W1 = (const float*)d_in[21];
  const float* att2b1 = (const float*)d_in[22];
  const float* att2W2 = (const float*)d_in[23];
  const float* att2b2 = (const float*)d_in[24];
  const float* g1W1   = (const float*)d_in[25];
  const float* g1b1   = (const float*)d_in[26];
  const float* g1W2   = (const float*)d_in[27];
  const float* g1b2   = (const float*)d_in[28];
  const float* g2W1   = (const float*)d_in[29];
  const float* g2b1   = (const float*)d_in[30];
  const float* g2W2   = (const float*)d_in[31];
  const float* g2b2   = (const float*)d_in[32];
  const float* outW1  = (const float*)d_in[33];
  const float* outb1  = (const float*)d_in[34];
  const float* outW2  = (const float*)d_in[35];
  const float* outb2  = (const float*)d_in[36];

  char* ws = (char*)d_ws;
  bf16_t* WL  = (bf16_t*)(ws + OFF_WL);
  bf16_t* WA  = (bf16_t*)(ws + OFF_WA);
  bf16_t* WV  = (bf16_t*)(ws + OFF_WV);
  float*  BL  = (float*)(ws + OFF_BL);
  float*  BA  = (float*)(ws + OFF_BA);
  float*  BV  = (float*)(ws + OFF_BV);
  bf16_t* A11 = (bf16_t*)(ws + OFF_A11);
  bf16_t* A12 = (bf16_t*)(ws + OFF_A12);
  bf16_t* A21 = (bf16_t*)(ws + OFF_A21);
  bf16_t* A22 = (bf16_t*)(ws + OFF_A22);
  bf16_t* G11 = (bf16_t*)(ws + OFF_G11);
  bf16_t* G12 = (bf16_t*)(ws + OFF_G12);
  bf16_t* G21 = (bf16_t*)(ws + OFF_G21);
  bf16_t* G22 = (bf16_t*)(ws + OFF_G22);
  bf16_t* OW1 = (bf16_t*)(ws + OFF_OW1);

  pack_gate<<<64, 256, 0, stream>>>(Wih_l, Whh_l, bih_l, bhh_l, WL, BL, DL, 448);
  pack_gate<<<64, 256, 0, stream>>>(Wih_a, Whh_a, bih_a, bhh_a, WA, BA, DA, 224);
  pack_gate<<<64, 256, 0, stream>>>(Wih_v, Whh_v, bih_v, bhh_v, WV, BV, DV, 192);
  pack_w<<<64, 256, 0, stream>>>(att1W1, A11, 256 * 768);
  pack_w<<<64, 256, 0, stream>>>(att1W2, A12, 768 * 256);
  pack_w<<<64, 256, 0, stream>>>(att2W1, A21, 256 * 768);
  pack_w<<<64, 256, 0, stream>>>(att2W2, A22, 256 * 256);
  pack_w<<<64, 256, 0, stream>>>(g1W1, G11, 256 * 1024);
  pack_w<<<64, 256, 0, stream>>>(g1W2, G12, 256 * 256);
  pack_w<<<64, 256, 0, stream>>>(g2W1, G21, 256 * 1024);
  pack_w<<<64, 256, 0, stream>>>(g2W2, G22, 256 * 256);
  pack_w<<<64, 256, 0, stream>>>(outW1, OW1, 256 * 640);

  (void)hipFuncSetAttribute((const void*)mfn_scan_kernel,
                            hipFuncAttributeMaxDynamicSharedMemorySize,
                            (int)SMEM_BYTES);
  mfn_scan_kernel<<<16, 256, SMEM_BYTES, stream>>>(
      x, WL, WA, WV, BL, BA, BV,
      A11, att1b1, A12, att1b2, A21, att2b1, A22, att2b2,
      G11, g1b1, G12, g1b2, G21, g2b1, G22, g2b2,
      OW1, outb1, outW2, outb2, (float*)d_out);
}
